// U_MLP_79156247265937
// MI455X (gfx1250) — matrix operand order repaired and checked by reading, not compiled, not run on hardware
//
#include <hip/hip_runtime.h>
#include <hip/hip_bf16.h>

#define B_ 4
#define S_ 4096
#define D_ 1024
#define E_ 8
#define H_ 256      // SUB_H
#define K_ 2048     // S/2 tokens kept per (expert,batch)
#define CAP_ 4.0f   // int(CAPACITY_FACTOR*B)
#define EPS_ 1e-6f

#define MT_ 64                    // tokens per block (4 M-tiles of 16)
#define XS_ST (D_ + 8)            // padded LDS row stride (elems): +16B -> +4 banks/row
#define HS_ST (H_ + 8)
#define XS_BYTES (MT_ * XS_ST * 2)
#define HS_BYTES (MT_ * HS_ST * 2)
#define LDS_BYTES (XS_BYTES + HS_BYTES + MT_ * 4)

typedef unsigned short u16;
typedef unsigned int   u32;
typedef __attribute__((ext_vector_type(16))) __bf16 bf16x16;
typedef __attribute__((ext_vector_type(8)))  float  f32x8;

union ABFrag { bf16x16 v; uint4 q[2]; };

__device__ __forceinline__ u16 f2bf(float f) {
    union { float f; u32 u; } c; c.f = f;
    u32 u = c.u;
    return (u16)((u + 0x7FFFu + ((u >> 16) & 1u)) >> 16);  // RNE
}

__device__ __forceinline__ float gelu_erf(float v) {
    return 0.5f * v * (1.0f + erff(v * 0.70710678118654752f));
}

// ---------------- zero output ----------------
__global__ void zero_kernel(float4* __restrict__ out, int n4) {
    int i = blockIdx.x * blockDim.x + threadIdx.x;
    if (i < n4) out[i] = make_float4(0.f, 0.f, 0.f, 0.f);
}

// ---------------- x fp32 -> bf16 ----------------
__global__ void conv_x_kernel(const float4* __restrict__ x, uint2* __restrict__ xb, int n4) {
    int i = blockIdx.x * blockDim.x + threadIdx.x;
    if (i >= n4) return;
    float4 v = x[i];
    uint2 o;
    o.x = (u32)f2bf(v.x) | ((u32)f2bf(v.y) << 16);
    o.y = (u32)f2bf(v.z) | ((u32)f2bf(v.w) << 16);
    xb[i] = o;
}

// W1 [E,D,H] fp32 -> W1t [E,H,D] bf16 (K-contiguous for WMMA B-fragments)
__global__ void conv_w1_kernel(const float* __restrict__ W1, u16* __restrict__ w1t) {
    int i = blockIdx.x * blockDim.x + threadIdx.x;
    if (i >= E_ * H_ * D_) return;
    int e = i / (H_ * D_);
    int r = i % (H_ * D_);
    int h = r / D_;
    int d = r % D_;
    w1t[i] = f2bf(W1[(size_t)e * D_ * H_ + (size_t)d * H_ + h]);
}

// W2 [E,H,D] fp32 -> W2t [E,D,H] bf16
__global__ void conv_w2_kernel(const float* __restrict__ W2, u16* __restrict__ w2t) {
    int i = blockIdx.x * blockDim.x + threadIdx.x;
    if (i >= E_ * D_ * H_) return;
    int e = i / (D_ * H_);
    int r = i % (D_ * H_);
    int d = r / H_;
    int h = r % H_;
    w2t[i] = f2bf(W2[(size_t)e * H_ * D_ + (size_t)h * D_ + d]);
}

// ---------------- gate: one wave32 per token ----------------
__global__ void gate_kernel(const float* __restrict__ x, const float* __restrict__ wg,
                            const float* __restrict__ bg, float* __restrict__ masked) {
    int wave = threadIdx.x >> 5, lane = threadIdx.x & 31;
    int tok  = blockIdx.x * 8 + wave;                 // 8 waves per block
    const float* xr = x + (size_t)tok * D_;
    float acc[E_];
    #pragma unroll
    for (int e = 0; e < E_; ++e) acc[e] = 0.f;
    for (int d = lane; d < D_; d += 32) {
        float xv = xr[d];
        const float* wr = wg + d * E_;
        #pragma unroll
        for (int e = 0; e < E_; ++e) acc[e] += xv * wr[e];
    }
    #pragma unroll
    for (int e = 0; e < E_; ++e) {
        #pragma unroll
        for (int off = 16; off > 0; off >>= 1)
            acc[e] += __shfl_xor(acc[e], off, 32);
    }
    if (lane == 0) {
        float mx = -1e30f;
        #pragma unroll
        for (int e = 0; e < E_; ++e) { acc[e] += bg[e]; mx = fmaxf(mx, acc[e]); }
        float p[E_], s = 0.f;
        #pragma unroll
        for (int e = 0; e < E_; ++e) { p[e] = __expf(acc[e] - mx); s += p[e]; }
        float inv = 1.0f / s;
        #pragma unroll
        for (int e = 0; e < E_; ++e) p[e] *= inv;
        int i1 = 0;
        #pragma unroll
        for (int e = 1; e < E_; ++e) if (p[e] > p[i1]) i1 = e;     // ties -> lower idx
        int i2 = (i1 == 0) ? 1 : 0;
        #pragma unroll
        for (int e = 0; e < E_; ++e) if (e != i1 && e != i2 && p[e] > p[i2]) i2 = e;
        float* mo = masked + (size_t)tok * E_;
        #pragma unroll
        for (int e = 0; e < E_; ++e) mo[e] = (e == i1 || e == i2) ? p[e] : 0.f;
    }
}

// ---------------- route = masked/(sum_b masked + eps) * capacity, laid out [E,B,S] ----
__global__ void route_kernel(const float* __restrict__ masked, float* __restrict__ route) {
    int i = blockIdx.x * blockDim.x + threadIdx.x;    // over S*E
    if (i >= S_ * E_) return;
    int s = i / E_, e = i % E_;
    float m[B_]; float den = EPS_;
    #pragma unroll
    for (int b = 0; b < B_; ++b) { m[b] = masked[((size_t)b * S_ + s) * E_ + e]; den += m[b]; }
    float scl = CAP_ / den;
    #pragma unroll
    for (int b = 0; b < B_; ++b)
        route[(size_t)(e * B_ + b) * S_ + s] = m[b] * scl;
}

// ---------------- top-K(=S/2) select per (e,b) via rank counting in LDS ----------------
__global__ void select_kernel(const float* __restrict__ route, int* __restrict__ sids) {
    __shared__ float v[S_];
    int eb = blockIdx.x;
    const float* r = route + (size_t)eb * S_;
    for (int i = threadIdx.x; i < S_; i += 256) v[i] = r[i];
    __syncthreads();
    const int T = S_ / 256;                            // 16 tokens per thread
    float mv[T]; int ms[T]; int rk[T];
    #pragma unroll
    for (int t = 0; t < T; ++t) { ms[t] = threadIdx.x + t * 256; mv[t] = v[ms[t]]; rk[t] = 0; }
    for (int j = 0; j < S_; ++j) {
        float vj = v[j];
        #pragma unroll
        for (int t = 0; t < T; ++t)
            rk[t] += (vj > mv[t]) || (vj == mv[t] && j < ms[t]);  // unique ranks
    }
    #pragma unroll
    for (int t = 0; t < T; ++t)
        if (rk[t] < K_) sids[(size_t)eb * K_ + rk[t]] = ms[t];
}

// ------- grouped FFN: 1 block = 64 gathered tokens (4 M-tiles), 8 waves, bf16 WMMA -----
__global__ __launch_bounds__(256) void ffn_kernel(
    const u16* __restrict__ xb, const u16* __restrict__ w1t, const u16* __restrict__ w2t,
    const float* __restrict__ b1, const float* __restrict__ b2,
    const int* __restrict__ sids, float* __restrict__ out)
{
    extern __shared__ char smem[];
    u16* xs  = (u16*)smem;                       // [64][XS_ST] bf16 gathered A tile
    u16* hs  = (u16*)(smem + XS_BYTES);          // [64][HS_ST] bf16 hidden tile
    int* sid = (int*)(smem + XS_BYTES + HS_BYTES);

    int blk  = blockIdx.x;                       // grid = E*B*(K/64) = 1024
    int tile = blk & 31;
    int b    = (blk >> 5) & 3;
    int e    = blk >> 7;

    if (threadIdx.x < MT_)
        sid[threadIdx.x] = sids[(size_t)(e * B_ + b) * K_ + tile * MT_ + threadIdx.x];
    __syncthreads();

    {   // gather 64 token rows (bf16): 4 threads/row, 512B contiguous each
        int r = threadIdx.x >> 2, c = threadIdx.x & 3;
        const uint4* src = (const uint4*)(xb + ((size_t)(b * S_ + sid[r])) * D_ + c * 256);
        uint4* dst = (uint4*)(xs + r * XS_ST + c * 256);
        #pragma unroll 8
        for (int i = 0; i < 32; ++i) dst[i] = src[i];
    }
    __syncthreads();

    int wave = threadIdx.x >> 5, lane = threadIdx.x & 31;
    int lm = lane & 15, lk = lane >> 4;

    // ---- phase 1: h = gelu(X @ W1 + b1); wave: 2 N-tiles x 4 M-tiles ----
    {
        f32x8 acc[4][2] = {};
        const u16* w1e = w1t + (size_t)e * H_ * D_;
        const u16* bpA = w1e + (size_t)(wave * 32 + lm) * D_ + lk * 16;
        const u16* bpB = bpA + 16 * D_;
        for (int k0 = 0; k0 < D_; k0 += 32) {
            ABFrag bA, bB;
            bA.q[0] = *(const uint4*)(bpA + k0);
            bA.q[1] = *(const uint4*)(bpA + k0 + 8);
            bB.q[0] = *(const uint4*)(bpB + k0);
            bB.q[1] = *(const uint4*)(bpB + k0 + 8);
            ABFrag a[4];
            #pragma unroll
            for (int mt = 0; mt < 4; ++mt) {       // preload all A frags: 1 clause, 1 wait
                const u16* ar = xs + (mt * 16 + lm) * XS_ST + k0 + lk * 16;
                a[mt].q[0] = *(const uint4*)(ar);
                a[mt].q[1] = *(const uint4*)(ar + 8);
            }
            #pragma unroll
            for (int mt = 0; mt < 4; ++mt) {       // 8 back-to-back WMMAs
                acc[mt][0] = __builtin_amdgcn_wmma_f32_16x16x32_bf16(false, a[mt].v, false, bA.v, (short)0, acc[mt][0], false, false);
                acc[mt][1] = __builtin_amdgcn_wmma_f32_16x16x32_bf16(false, a[mt].v, false, bB.v, (short)0, acc[mt][1], false, false);
            }
        }
        int col0 = wave * 32 + lm;
        float bias0 = b1[e * H_ + col0], bias1 = b1[e * H_ + col0 + 16];
        #pragma unroll
        for (int mt = 0; mt < 4; ++mt) {
            #pragma unroll
            for (int j = 0; j < 8; ++j) {
                int m = mt * 16 + j + 8 * lk;          // C/D layout: row = j + 8*(lane>=16)
                hs[m * HS_ST + col0]      = f2bf(gelu_erf(acc[mt][0][j] + bias0));
                hs[m * HS_ST + col0 + 16] = f2bf(gelu_erf(acc[mt][1][j] + bias1));
            }
        }
    }
    __syncthreads();

    // ---- phase 2: o = h @ W2 + b2; wave covers 128 cols in 4 chunks of 2 N-tiles ----
    const u16* w2e = w2t + (size_t)e * D_ * H_;
    for (int ng = 0; ng < 4; ++ng) {
        int n0 = wave * 128 + ng * 32;
        f32x8 acc[4][2] = {};
        const u16* bpA = w2e + (size_t)(n0 + lm) * H_ + lk * 16;
        const u16* bpB = bpA + 16 * H_;
        for (int k0 = 0; k0 < H_; k0 += 32) {
            ABFrag bA, bB;
            bA.q[0] = *(const uint4*)(bpA + k0);
            bA.q[1] = *(const uint4*)(bpA + k0 + 8);
            bB.q[0] = *(const uint4*)(bpB + k0);
            bB.q[1] = *(const uint4*)(bpB + k0 + 8);
            ABFrag a[4];
            #pragma unroll
            for (int mt = 0; mt < 4; ++mt) {
                const u16* ar = hs + (mt * 16 + lm) * HS_ST + k0 + lk * 16;
                a[mt].q[0] = *(const uint4*)(ar);
                a[mt].q[1] = *(const uint4*)(ar + 8);
            }
            #pragma unroll
            for (int mt = 0; mt < 4; ++mt) {
                acc[mt][0] = __builtin_amdgcn_wmma_f32_16x16x32_bf16(false, a[mt].v, false, bA.v, (short)0, acc[mt][0], false, false);
                acc[mt][1] = __builtin_amdgcn_wmma_f32_16x16x32_bf16(false, a[mt].v, false, bB.v, (short)0, acc[mt][1], false, false);
            }
        }
        // scatter-add this chunk (f32 L2 atomics), bias folded in
        float biasA = b2[e * D_ + n0 + lm];
        float biasB = b2[e * D_ + n0 + 16 + lm];
        #pragma unroll
        for (int mt = 0; mt < 4; ++mt) {
            #pragma unroll
            for (int j = 0; j < 8; ++j) {
                int m = mt * 16 + j + 8 * lk;
                float* orow = out + ((size_t)(b * S_ + sid[m])) * D_;
                atomicAdd(orow + n0 + lm,      acc[mt][0][j] + biasA);
                atomicAdd(orow + n0 + 16 + lm, acc[mt][1][j] + biasB);
            }
        }
    }
}

extern "C" void kernel_launch(void* const* d_in, const int* in_sizes, int n_in,
                              void* d_out, int out_size, void* d_ws, size_t ws_size,
                              hipStream_t stream) {
    const float* x   = (const float*)d_in[0];
    const float* wgw = (const float*)d_in[1];
    const float* wgb = (const float*)d_in[2];
    const float* W1  = (const float*)d_in[3];
    const float* b1  = (const float*)d_in[4];
    const float* W2  = (const float*)d_in[5];
    const float* b2  = (const float*)d_in[6];
    float* out = (float*)d_out;

    char* ws = (char*)d_ws;
    size_t off = 0;
    auto carve = [&](size_t bytes) {
        char* p = ws + off;
        off = (off + bytes + 255) & ~(size_t)255;
        return p;
    };
    u16*   xbf    = (u16*)  carve((size_t)B_ * S_ * D_ * 2);   // 32 MB
    u16*   w1t    = (u16*)  carve((size_t)E_ * H_ * D_ * 2);   // 4 MB
    u16*   w2t    = (u16*)  carve((size_t)E_ * D_ * H_ * 2);   // 4 MB
    float* masked = (float*)carve((size_t)B_ * S_ * E_ * 4);   // 512 KB
    float* route  = (float*)carve((size_t)E_ * B_ * S_ * 4);   // 512 KB
    int*   sids   = (int*)  carve((size_t)E_ * B_ * K_ * 4);   // 256 KB

    // allow >64KB dynamic LDS for the FFN kernel (not a stream op; capture-safe)
    static bool attr_done = false;
    if (!attr_done) {
        (void)hipFuncSetAttribute((const void*)ffn_kernel,
                                  hipFuncAttributeMaxDynamicSharedMemorySize, LDS_BYTES);
        attr_done = true;
    }

    int n4 = B_ * S_ * D_ / 4;
    zero_kernel  <<<(n4 + 255) / 256, 256, 0, stream>>>((float4*)out, n4);
    conv_x_kernel<<<(n4 + 255) / 256, 256, 0, stream>>>((const float4*)x, (uint2*)xbf, n4);
    int nw = E_ * H_ * D_;
    conv_w1_kernel<<<(nw + 255) / 256, 256, 0, stream>>>(W1, w1t);
    conv_w2_kernel<<<(nw + 255) / 256, 256, 0, stream>>>(W2, w2t);

    gate_kernel  <<<B_ * S_ / 8, 256, 0, stream>>>(x, wgw, wgb, masked);
    route_kernel <<<(S_ * E_ + 255) / 256, 256, 0, stream>>>(masked, route);
    select_kernel<<<E_ * B_, 256, 0, stream>>>(route, sids);

    ffn_kernel<<<E_ * B_ * (K_ / MT_), 256, LDS_BYTES, stream>>>(
        xbf, w1t, w2t, b1, b2, sids, out);
}
